// Stage2ModelPT_22857815949891
// MI455X (gfx1250) — matrix operand order repaired and checked by reading, not compiled, not run on hardware
//
#include <hip/hip_runtime.h>

typedef _Float16 v16h __attribute__((ext_vector_type(16)));
typedef _Float16 v8h  __attribute__((ext_vector_type(8)));
typedef float    v8f  __attribute__((ext_vector_type(8)));

#define NN 4096
#define KK 12288          // stacked K: [Wsv^T | Wdcv^T | L]
#define TT 16

__device__ __forceinline__ float sp_f(float x) {            // softplus
  return (x > 20.f) ? x : log1pf(expf(x));
}
__device__ __forceinline__ float sig_f(float x) {           // sigmoid
  return 1.f / (1.f + expf(-x));
}

// ---------------- precompute: column sums of T_e ----------------
__global__ __launch_bounds__(256) void k_colsum(const float* __restrict__ Te,
                                                float* __restrict__ degsum) {
  int i = blockIdx.x * 256 + threadIdx.x;
  float s = 0.f;
  for (int j = 0; j < NN; ++j) s += Te[(size_t)j * NN + i];
  degsum[i] = s;
}

// ---------------- precompute: add row sums of T_e ----------------
__global__ __launch_bounds__(256) void k_rowsum(const float* __restrict__ Te,
                                                float* __restrict__ degsum) {
  __shared__ float red[256];
  int r = blockIdx.x;
  float s = 0.f;
  for (int j = threadIdx.x; j < NN; j += 256) s += Te[(size_t)r * NN + j];
  red[threadIdx.x] = s;
  __syncthreads();
  for (int o = 128; o > 0; o >>= 1) {
    if (threadIdx.x < o) red[threadIdx.x] += red[threadIdx.x + o];
    __syncthreads();
  }
  if (threadIdx.x == 0) degsum[r] += red[0];
}

// ---------------- precompute: build stacked transposed f16 matrix ----------
// ST[i, k] row-major, i=dest (4096 rows), k in [0,12288):
//   k <  4096 : Wsv[k,i]  = Tsv[k,i]*softplus(Wsv_raw[k,i])
//   k <  8192 : Wdcv[k',i]
//   else      : L[k',i] (symmetric; diag = G*Te[i,i] - 0.5*G*(rowsum+colsum))
__global__ __launch_bounds__(256) void k_build_st(
    const float* __restrict__ Te, const float* __restrict__ Tsv,
    const float* __restrict__ Tdcv, const float* __restrict__ Wsv_raw,
    const float* __restrict__ Wdcv_raw, const float* __restrict__ G_raw,
    const float* __restrict__ degsum, _Float16* __restrict__ ST) {
  __shared__ float tile[32][33];
  const int tx = threadIdx.x & 31;
  const int ty = threadIdx.x >> 5;           // 0..7
  const int k0g = blockIdx.x * 32;           // global stacked k
  const int i0  = blockIdx.y * 32;
  const int seg = k0g >> 12;                 // 0,1,2
  const int k0  = k0g & 4095;                // seg-local k
#pragma unroll
  for (int rr = 0; rr < 4; ++rr) {
    int kk = k0 + ty + rr * 8;
    int ii = i0 + tx;
    size_t idx = (size_t)kk * NN + ii;       // coalesced read [k][i]
    float v;
    if (seg == 0)      v = Tsv[idx]  * sp_f(Wsv_raw[idx]);
    else if (seg == 1) v = Tdcv[idx] * sp_f(Wdcv_raw[idx]);
    else               v = Te[idx];
    tile[ty + rr * 8][tx] = v;
  }
  __syncthreads();
  const float G = sp_f(G_raw[0]);
#pragma unroll
  for (int rr = 0; rr < 4; ++rr) {
    int irow = ty + rr * 8;
    int i    = i0 + irow;
    int kloc = k0 + tx;
    float v = tile[tx][irow];                // element (k=kloc, i)
    if (seg == 2) {
      float v2 = Te[(size_t)i * NN + kloc];  // coalesced read of Te[i,k]
      float wv = 0.5f * G * (v + v2);
      if (i == kloc) wv = G * v - 0.5f * G * degsum[i];
      v = wv;
    }
    ST[(size_t)i * KK + k0g + tx] = (_Float16)v;   // coalesced write
  }
}

// ---------------- precompute: I_stim[t][i] = (b @ stim_t)[i] ----------------
__global__ __launch_bounds__(256) void k_istim(const float* __restrict__ b,
                                               const float* __restrict__ stim,
                                               float* __restrict__ Istim) {
  __shared__ float ss[TT * 64];
  for (int idx = threadIdx.x; idx < TT * 64; idx += 256) ss[idx] = stim[idx];
  __syncthreads();
  int i = blockIdx.x * 256 + threadIdx.x;
  float acc[TT];
#pragma unroll
  for (int t = 0; t < TT; ++t) acc[t] = 0.f;
  for (int d = 0; d < 64; ++d) {
    float bv = b[(size_t)i * 64 + d];
#pragma unroll
    for (int t = 0; t < TT; ++t) acc[t] += bv * ss[t * 64 + d];
  }
#pragma unroll
  for (int t = 0; t < TT; ++t) Istim[t * NN + i] = acc[t];
}

// ---------------- zero the 16 x KK f16 A-buffer ----------------
__global__ __launch_bounds__(256) void k_zero_abuf(unsigned int* __restrict__ p) {
  int idx = blockIdx.x * 256 + threadIdx.x;
  if (idx < 16 * KK / 2) p[idx] = 0u;
}

// ---------------- initial state / A rows for step 0 ----------------
__global__ __launch_bounds__(256) void k_prep(
    const float* __restrict__ u0, const float* __restrict__ s_sv0,
    const float* __restrict__ s_dcv0, const float* __restrict__ gating,
    const float* __restrict__ a_sv_raw, const float* __restrict__ tau_sv_raw,
    const float* __restrict__ a_dcv_raw, const float* __restrict__ tau_dcv_raw,
    const float* __restrict__ E_sv, float* __restrict__ u_cur,
    float* __restrict__ s_sv, float* __restrict__ s_dcv,
    _Float16* __restrict__ Abuf) {
  int i = blockIdx.x * 256 + threadIdx.x;
  float u = u0[i];
  u_cur[i] = u;
  float phi = gating[i] * sig_f(u);
  float a0 = sp_f(a_sv_raw[0]),  a1 = sp_f(a_sv_raw[1]);
  float g0 = expf(-0.01f / (sp_f(tau_sv_raw[0]) + 1e-4f + 1e-12f));
  float g1 = expf(-0.01f / (sp_f(tau_sv_raw[1]) + 1e-4f + 1e-12f));
  float s0 = g0 * s_sv0[i * 2 + 0] + phi;
  float s1 = g1 * s_sv0[i * 2 + 1] + phi;
  s_sv[i] = s0; s_sv[NN + i] = s1;
  float p = a0 * s0 + a1 * s1;
  float q = E_sv[i] * p;
  float b0 = sp_f(a_dcv_raw[0]), b1 = sp_f(a_dcv_raw[1]);
  float h0 = expf(-0.01f / (sp_f(tau_dcv_raw[0]) + 1e-4f + 1e-12f));
  float h1 = expf(-0.01f / (sp_f(tau_dcv_raw[1]) + 1e-4f + 1e-12f));
  float d0 = h0 * s_dcv0[i * 2 + 0] + phi;
  float d1 = h1 * s_dcv0[i * 2 + 1] + phi;
  s_dcv[i] = d0; s_dcv[NN + i] = d1;
  float pd = b0 * d0 + b1 * d1;
  Abuf[i]                 = (_Float16)p;
  Abuf[KK + i]            = (_Float16)q;
  Abuf[2 * KK + NN + i]   = (_Float16)pd;
  Abuf[3 * KK + 2 * NN + i] = (_Float16)u;
}

// ---------------- per-step WMMA GEMM: outvec[4][NN] = A(16xKK) x ST^T -------
// Block = 8 waves, one 16-destination tile; each wave covers 48 K-chunks of 32.
__global__ __launch_bounds__(256) void k_gemm(const _Float16* __restrict__ ST,
                                              const _Float16* __restrict__ Abuf,
                                              float* __restrict__ outvec) {
  __shared__ float red[512];
  const int tid  = threadIdx.x;
  const int lane = tid & 31;
  const int w    = tid >> 5;
  const int i0   = blockIdx.x * 16;
  const int m    = lane & 15;
  const int hi   = lane >> 4;
  // A 16x32 f16 layout: lane holds row m, 16 contiguous K at offset hi*16
  const _Float16* aBase = Abuf + (size_t)m * KK + hi * 16;
  // B 32x16 f16 layout: lane holds column n=m, 16 contiguous K at offset hi*16
  const _Float16* bBase = ST + (size_t)(i0 + m) * KK + hi * 16;
  v8f acc = {};
  const int kc0 = w * 48;
  for (int kc = 0; kc < 48; ++kc) {
    const int kb = (kc0 + kc) * 32;
    v8h alo = *(const v8h*)(aBase + kb);
    v8h ahi = *(const v8h*)(aBase + kb + 8);
    v16h a = __builtin_shufflevector(alo, ahi, 0, 1, 2, 3, 4, 5, 6, 7,
                                     8, 9, 10, 11, 12, 13, 14, 15);
    v16h bm = *(const v16h*)(bBase + kb);
    acc = __builtin_amdgcn_wmma_f32_16x16x32_f16(false, a, false, bm,
                                                 (short)0, acc, false, false);
  }
  // D f32 layout: lanes 0-15, VGPR r -> (M=r, N=lane); channels are M=0..3
  if (lane < 16) {
#pragma unroll
    for (int r = 0; r < 4; ++r) red[w * 64 + r * 16 + lane] = acc[r];
  }
  __syncthreads();
  if (tid < 64) {
    float s = 0.f;
#pragma unroll
    for (int ww = 0; ww < 8; ++ww) s += red[ww * 64 + tid];
    outvec[(tid >> 4) * NN + i0 + (tid & 15)] = s;
  }
}

// ---------------- per-step pointwise update + next-step A rows ----------------
__global__ __launch_bounds__(256) void k_update(
    const float* __restrict__ outvec, float* __restrict__ u_cur,
    float* __restrict__ s_sv, float* __restrict__ s_dcv,
    _Float16* __restrict__ Abuf, float* __restrict__ out,
    const float* __restrict__ I0, const float* __restrict__ lam_raw,
    const float* __restrict__ a_sv_raw, const float* __restrict__ tau_sv_raw,
    const float* __restrict__ a_dcv_raw, const float* __restrict__ tau_dcv_raw,
    const float* __restrict__ E_sv, const float* __restrict__ gating,
    const float* __restrict__ Istim, int t) {
  int i = blockIdx.x * 256 + threadIdx.x;
  float u  = u_cur[i];
  float o0 = outvec[i];            // g_sv
  float o1 = outvec[NN + i];       // sum Wsv^T (E.*p)
  float o2 = outvec[2 * NN + i];   // g_dcv
  float o3 = outvec[3 * NN + i];   // I_gap
  float lam = 0.9999f * sig_f(lam_raw[i]);
  float rhs = I0[i] + o3 + (o1 - o0 * u) - o2 * u + Istim[t * NN + i];
  float un = (1.f - lam) * u + lam * rhs;
  un = fminf(fmaxf(un, -35.f), 35.f);
  out[t * NN + i] = un;
  u_cur[i] = un;
  if (t + 1 < TT) {
    float phi = gating[(t + 1) * NN + i] * sig_f(un);
    float a0 = sp_f(a_sv_raw[0]),  a1 = sp_f(a_sv_raw[1]);
    float g0 = expf(-0.01f / (sp_f(tau_sv_raw[0]) + 1e-4f + 1e-12f));
    float g1 = expf(-0.01f / (sp_f(tau_sv_raw[1]) + 1e-4f + 1e-12f));
    float s0 = g0 * s_sv[i] + phi;
    float s1 = g1 * s_sv[NN + i] + phi;
    s_sv[i] = s0; s_sv[NN + i] = s1;
    float p = a0 * s0 + a1 * s1;
    float q = E_sv[i] * p;
    float b0 = sp_f(a_dcv_raw[0]), b1 = sp_f(a_dcv_raw[1]);
    float h0 = expf(-0.01f / (sp_f(tau_dcv_raw[0]) + 1e-4f + 1e-12f));
    float h1 = expf(-0.01f / (sp_f(tau_dcv_raw[1]) + 1e-4f + 1e-12f));
    float d0 = h0 * s_dcv[i] + phi;
    float d1 = h1 * s_dcv[NN + i] + phi;
    s_dcv[i] = d0; s_dcv[NN + i] = d1;
    float pd = b0 * d0 + b1 * d1;
    Abuf[i]                   = (_Float16)p;
    Abuf[KK + i]              = (_Float16)q;
    Abuf[2 * KK + NN + i]     = (_Float16)pd;
    Abuf[3 * KK + 2 * NN + i] = (_Float16)un;
  }
}

extern "C" void kernel_launch(void* const* d_in, const int* in_sizes, int n_in,
                              void* d_out, int out_size, void* d_ws, size_t ws_size,
                              hipStream_t stream) {
  (void)in_sizes; (void)n_in; (void)out_size; (void)ws_size;
  const float* Te       = (const float*)d_in[0];
  const float* Tsv      = (const float*)d_in[1];
  const float* Tdcv     = (const float*)d_in[2];
  const float* Wsv_raw  = (const float*)d_in[3];
  const float* Wdcv_raw = (const float*)d_in[4];
  const float* G_raw    = (const float*)d_in[5];
  const float* lam_raw  = (const float*)d_in[6];
  const float* a_sv_raw   = (const float*)d_in[7];
  const float* tau_sv_raw = (const float*)d_in[8];
  const float* a_dcv_raw  = (const float*)d_in[9];
  const float* tau_dcv_raw = (const float*)d_in[10];
  const float* I0   = (const float*)d_in[11];
  const float* b    = (const float*)d_in[12];
  const float* E_sv = (const float*)d_in[13];
  const float* u0   = (const float*)d_in[14];
  const float* s_sv0 = (const float*)d_in[15];
  const float* s_dcv0 = (const float*)d_in[16];
  const float* gating = (const float*)d_in[17];
  const float* stim   = (const float*)d_in[18];
  float* out = (float*)d_out;

  char* ws = (char*)d_ws;
  const size_t ST_B = (size_t)NN * KK * 2;      // 96 MB stacked f16 matrix
  const size_t AB_B = (size_t)16 * KK * 2;      // 384 KB A buffer
  _Float16* ST   = (_Float16*)ws;
  _Float16* Abuf = (_Float16*)(ws + ST_B);
  float* u_cur  = (float*)(ws + ST_B + AB_B);
  float* s_sv   = u_cur + NN;
  float* s_dcv  = s_sv + 2 * NN;
  float* Istim  = s_dcv + 2 * NN;
  float* outvec = Istim + TT * NN;
  float* degsum = outvec + 4 * NN;

  k_colsum<<<NN / 256, 256, 0, stream>>>(Te, degsum);
  k_rowsum<<<NN, 256, 0, stream>>>(Te, degsum);
  k_build_st<<<dim3(KK / 32, NN / 32), 256, 0, stream>>>(
      Te, Tsv, Tdcv, Wsv_raw, Wdcv_raw, G_raw, degsum, ST);
  k_istim<<<NN / 256, 256, 0, stream>>>(b, stim, Istim);
  k_zero_abuf<<<(16 * KK / 2 + 255) / 256, 256, 0, stream>>>((unsigned int*)Abuf);
  k_prep<<<NN / 256, 256, 0, stream>>>(u0, s_sv0, s_dcv0, gating, a_sv_raw,
                                       tau_sv_raw, a_dcv_raw, tau_dcv_raw, E_sv,
                                       u_cur, s_sv, s_dcv, Abuf);
  for (int t = 0; t < TT; ++t) {
    k_gemm<<<NN / 16, 256, 0, stream>>>(ST, Abuf, outvec);
    k_update<<<NN / 256, 256, 0, stream>>>(outvec, u_cur, s_sv, s_dcv, Abuf, out,
                                           I0, lam_raw, a_sv_raw, tau_sv_raw,
                                           a_dcv_raw, tau_dcv_raw, E_sv, gating,
                                           Istim, t);
  }
}
